// RelationGraphAttention_21534966022950
// MI455X (gfx1250) — compile-verified, matrix operand order checked by reading, not run on hardware
//
#include <hip/hip_runtime.h>
#include <hip/hip_bf16.h>

// MI455X / gfx1250 implementation. wave32 everywhere. All dense GEMMs use
// v_wmma_f32_16x16x32_bf16 (bf16 A/B, f32 accumulate). Weights are
// pre-converted to bf16 once per launch so B-fragment loads are pure b64
// loads with no cvt chain in the hot loops.

typedef __attribute__((ext_vector_type(16))) __bf16 v16bf;
typedef __attribute__((ext_vector_type(8)))  float  v8f;

#define HID 128
#define EDIM 32
#define NHEAD 4
#define HD 32

// ---- fragment helpers -----------------------------------------------------
// 16-bit A/B fragment K order per CDNA5 ISA 7.12.2:
//   half = lane>>4 ; element j of the lane's 16 bf16 values holds
//   K = (j&7) | (half<<3) | ((j>>3)<<4)   (covers K=0..31 across both halves)
__device__ __forceinline__ int kperm(int half, int j) {
  return (j & 7) | (half << 3) | ((j >> 3) << 4);
}

__device__ __forceinline__ v16bf frag_from_row(const float* row, int half, int kbase) {
  v16bf f;
#pragma unroll
  for (int j = 0; j < 16; ++j) f[j] = (__bf16)row[kbase + kperm(half, j)];
  return f;
}

// bf16 source: two contiguous 8-element (16B) chunks per fragment, no cvt.
__device__ __forceinline__ v16bf frag_from_row_bf(const __bf16* row, int half, int kbase) {
  v16bf f;
#pragma unroll
  for (int j = 0; j < 16; ++j) f[j] = row[kbase + kperm(half, j)];
  return f;
}

__device__ __forceinline__ v8f wmma_bf16(v16bf a, v16bf b, v8f c) {
  return __builtin_amdgcn_wmma_f32_16x16x32_bf16(false, a, false, b, (short)0, c, false, false);
}

// order-preserving float<->uint encoding for atomicMax-based segment max
__device__ __forceinline__ unsigned enc_f(float f) {
  unsigned u = __float_as_uint(f);
  return (u >> 31) ? ~u : (u | 0x80000000u);
}
__device__ __forceinline__ float dec_f(unsigned e) {
  return __uint_as_float((e >> 31) ? (e & 0x7fffffffu) : ~e);
}

// bf16 weight workspace offsets (elements)
#define OFF_WSRC   0
#define OFF_WDST   16384            // 128*128
#define OFF_WEDGE  (16384 * 2)      // +128*128
#define OFF_WMSG   (OFF_WEDGE + 4096)   // +128*32
#define OFF_WMERGE (OFF_WMSG + 1024)    // +32*32
#define OFF_WOUT   (OFF_WMERGE + 4096)  // +128*32
#define WB_TOTAL   (OFF_WOUT + 32768)   // +128*256 = 74752 elements

// ---- kernel 0: one-shot f32 -> bf16 weight conversion ---------------------
__global__ void cvt_weights_kernel(const float* __restrict__ Wsrc, const float* __restrict__ Wdst,
                                   const float* __restrict__ Wedge, const float* __restrict__ Wmsg,
                                   const float* __restrict__ Wmerge, const float* __restrict__ Wout,
                                   __bf16* __restrict__ wb) {
  int t = blockIdx.x * blockDim.x + threadIdx.x;
  if (t >= WB_TOTAL) return;
  float v;
  if      (t < OFF_WDST)   v = Wsrc[t - OFF_WSRC];
  else if (t < OFF_WEDGE)  v = Wdst[t - OFF_WDST];
  else if (t < OFF_WMSG)   v = Wedge[t - OFF_WEDGE];
  else if (t < OFF_WMERGE) v = Wmsg[t - OFF_WMSG];
  else if (t < OFF_WOUT)   v = Wmerge[t - OFF_WMERGE];
  else                     v = Wout[t - OFF_WOUT];
  wb[t] = (__bf16)v;
}

// ---- kernel 1: per-node projection  Out[r,:] = X[r,:] @ W.T + b -----------
// Wb is [128,128] row-major bf16 (out,in). Wave handles 16 rows x 128 cols.
__global__ void proj_kernel(const float* __restrict__ X, const __bf16* __restrict__ Wb,
                            const float* __restrict__ b, float* __restrict__ Out, int nrows) {
  int wave = (blockIdx.x * blockDim.x + threadIdx.x) >> 5;
  int lane = threadIdx.x & 31;
  int row0 = wave * 16;
  if (row0 >= nrows) return;
  int m = lane & 15, half = lane >> 4;
  const float* arow = X + (size_t)(row0 + m) * HID;
  v16bf A[4];
#pragma unroll
  for (int kt = 0; kt < 4; ++kt) A[kt] = frag_from_row(arow, half, kt * 32);
#pragma unroll
  for (int nt = 0; nt < 8; ++nt) {
    int n = (lane & 15) + nt * 16;
    const __bf16* brow = Wb + (size_t)n * HID;   // col n of W.T == row n of W
    v8f c = {};
#pragma unroll
    for (int kt = 0; kt < 4; ++kt)
      c = wmma_bf16(A[kt], frag_from_row_bf(brow, half, kt * 32), c);
    float bias = b[n];
#pragma unroll
    for (int r = 0; r < 8; ++r) {
      int mm = r + half * 8;                  // C layout: M=r (+8 for hi half)
      Out[(size_t)(row0 + mm) * HID + n] = c[r] + bias;
    }
  }
}

// ---- shared helper: e_feat[16][128] tile into LDS via WMMA ---------------
__device__ __forceinline__ void efeat_to_lds(const float* __restrict__ edge_attr,
                                             const __bf16* __restrict__ WedgeB,
                                             const float* __restrict__ bedge,
                                             int e_base, int lane, float* efeat) {
  int m = lane & 15, half = lane >> 4;
  v16bf A = frag_from_row(edge_attr + (size_t)(e_base + m) * EDIM, half, 0);
#pragma unroll
  for (int nt = 0; nt < 8; ++nt) {
    int n = (lane & 15) + nt * 16;
    v16bf B = frag_from_row_bf(WedgeB + (size_t)n * EDIM, half, 0);
    v8f c = {};
    c = wmma_bf16(A, B, c);
    float bias = bedge[n];
#pragma unroll
    for (int r = 0; r < 8; ++r) efeat[(r + half * 8) * HID + n] = c[r] + bias;
  }
}

#define WPB 4   // waves per block for edge kernels (4 * 8KB LDS tiles)

// ---- kernel 3: attention scores + segment max ----------------------------
__global__ void edge_scores_kernel(const float* __restrict__ edge_attr,
                                   const __bf16* __restrict__ WedgeB, const float* __restrict__ bedge,
                                   const float* __restrict__ Wattn, const float* __restrict__ battn,
                                   const int* __restrict__ src_idx, const int* __restrict__ dst_idx,
                                   const float* __restrict__ SrcP, const float* __restrict__ DstP,
                                   float* __restrict__ scores, unsigned* __restrict__ smax, int E_) {
  __shared__ float smem[WPB * 16 * HID];
  __shared__ int sidx[WPB * 16], didx[WPB * 16];
  int wv = threadIdx.x >> 5, lane = threadIdx.x & 31;
  int tile = blockIdx.x * WPB + wv;
  if (tile * 16 >= E_) return;
  int e_base = tile * 16;
  float* efeat = smem + wv * 16 * HID;
  int* si = sidx + wv * 16;
  int* di = didx + wv * 16;
  if (lane < 16) { si[lane] = src_idx[e_base + lane]; di[lane] = dst_idx[e_base + lane]; }
  efeat_to_lds(edge_attr, WedgeB, bedge, e_base, lane, efeat);
  // 64 (edge,head) pairs per wave; lane handles p = lane and lane+32
  float b0 = battn[0];
#pragma unroll
  for (int pp = 0; pp < 2; ++pp) {
    int p = lane + pp * 32;
    int el = p >> 2, h = p & 3;
    const float* sp = SrcP + (size_t)si[el] * HID + h * HD;
    const float* dp = DstP + (size_t)di[el] * HID + h * HD;
    const float* ep = efeat + el * HID + h * HD;
    float acc = b0;
#pragma unroll 4
    for (int d = 0; d < HD; ++d) {
      acc += tanhf(sp[d]) * Wattn[d]
           + tanhf(dp[d]) * Wattn[HD + d]
           + tanhf(ep[d]) * Wattn[2 * HD + d];
    }
    scores[(size_t)(e_base + el) * NHEAD + h] = acc;
    atomicMax(&smax[(size_t)di[el] * NHEAD + h], enc_f(acc));
  }
}

// ---- kernel 4: exp(score - max) + segment sum ----------------------------
__global__ void edge_exp_kernel(const int* __restrict__ dst_idx, float* __restrict__ scores,
                                const unsigned* __restrict__ smax, float* __restrict__ ssum,
                                int total) {
  int t = blockIdx.x * blockDim.x + threadIdx.x;
  if (t >= total) return;
  int e = t >> 2, h = t & 3;
  int d = dst_idx[e];
  float mx = dec_f(smax[(size_t)d * NHEAD + h]);
  float ex = __expf(scores[t] - mx);
  scores[t] = ex;                       // overwrite with exp'ed score
  atomicAdd(&ssum[(size_t)d * NHEAD + h], ex);
}

// ---- kernel 5: msg WMMA + alpha-weighted scatter (head-mean folded) -------
__global__ void edge_aggregate_kernel(const float* __restrict__ edge_attr,
                                      const __bf16* __restrict__ WedgeB, const float* __restrict__ bedge,
                                      const __bf16* __restrict__ WmsgB, const float* __restrict__ bmsg,
                                      const int* __restrict__ src_idx, const int* __restrict__ dst_idx,
                                      const float* __restrict__ SrcP,
                                      const float* __restrict__ scores, const float* __restrict__ ssum,
                                      float* __restrict__ agg, int E_) {
  __shared__ float smem[WPB * 16 * HID];
  __shared__ int sidx[WPB * 16], didx[WPB * 16];
  __shared__ float alph[WPB * 64];
  int wv = threadIdx.x >> 5, lane = threadIdx.x & 31;
  int tile = blockIdx.x * WPB + wv;
  if (tile * 16 >= E_) return;
  int e_base = tile * 16;
  float* efeat = smem + wv * 16 * HID;
  int* si = sidx + wv * 16;
  int* di = didx + wv * 16;
  float* al = alph + wv * 64;
  if (lane < 16) { si[lane] = src_idx[e_base + lane]; di[lane] = dst_idx[e_base + lane]; }
  efeat_to_lds(edge_attr, WedgeB, bedge, e_base, lane, efeat);
  // alpha = exp / segsum for the wave's 64 (edge,head) pairs
#pragma unroll
  for (int pp = 0; pp < 2; ++pp) {
    int p = lane + pp * 32;
    int el = p >> 2, h = p & 3;
    al[el * NHEAD + h] =
        scores[(size_t)(e_base + el) * NHEAD + h] / ssum[(size_t)di[el] * NHEAD + h];
  }
  // efeat += gathered SrcP row (src_feat + e_feat)
  {
    int el = lane >> 1;
    int c0 = (lane & 1) * 64;
    const float* sp = SrcP + (size_t)si[el] * HID + c0;
    float* ef = efeat + el * HID + c0;
#pragma unroll 8
    for (int j = 0; j < 64; ++j) ef[j] += sp[j];
  }
  // per-head msg = (src+e) @ Wmsg.T + bmsg  (M=16,N=32,K=32), then scatter
  int m = lane & 15, half = lane >> 4;
#pragma unroll
  for (int h = 0; h < NHEAD; ++h) {
    v16bf Ah = frag_from_row(efeat + m * HID + h * HD, half, 0);
#pragma unroll
    for (int sub = 0; sub < 2; ++sub) {
      int dl = (lane & 15) + sub * 16;            // output dim within head
      v16bf B = frag_from_row_bf(WmsgB + (size_t)dl * HD, half, 0);
      v8f c = {};
      c = wmma_bf16(Ah, B, c);
      float bias = bmsg[dl];
#pragma unroll
      for (int r = 0; r < 8; ++r) {
        int mm = r + half * 8;
        float val = (c[r] + bias) * al[mm * NHEAD + h] * 0.25f;  // mean over heads
        atomicAdd(&agg[(size_t)di[mm] * HD + dl], val);
      }
    }
  }
}

// ---- kernel 6: Wmerge + Wout + residual + LayerNorm -----------------------
__global__ void finalize_kernel(const float* __restrict__ dst_x, const float* __restrict__ agg,
                                const __bf16* __restrict__ WmergeB, const float* __restrict__ bmerge,
                                const __bf16* __restrict__ WoutB, const float* __restrict__ bout,
                                const float* __restrict__ gamma, const float* __restrict__ beta,
                                float* __restrict__ out, int N_) {
  __shared__ float smem[2 * 2 * 16 * HID];   // 2 waves x (aggm tile + row tile)
  int wv = threadIdx.x >> 5, lane = threadIdx.x & 31;
  int tile = blockIdx.x * 2 + wv;
  if (tile * 16 >= N_) return;
  int row0 = tile * 16;
  float* aggm = smem + wv * 2 * 16 * HID;
  float* rows = aggm + 16 * HID;
  int m = lane & 15, half = lane >> 4;
  // stage 1: aggm = agg_mean @ Wmerge.T + bmerge   (M=16,N=128,K=32)
  {
    v16bf Aa = frag_from_row(agg + (size_t)(row0 + m) * HD, half, 0);
#pragma unroll
    for (int nt = 0; nt < 8; ++nt) {
      int n = (lane & 15) + nt * 16;
      v16bf B = frag_from_row_bf(WmergeB + (size_t)n * HD, half, 0);
      v8f c = {};
      c = wmma_bf16(Aa, B, c);
      float bias = bmerge[n];
#pragma unroll
      for (int r = 0; r < 8; ++r) aggm[(r + half * 8) * HID + n] = c[r] + bias;
    }
  }
  // stage 2: updated = [dst_x | aggm] @ Wout.T + bout ; rows = dst_x + updated
  {
    v16bf A[8];
    const float* xr = dst_x + (size_t)(row0 + m) * HID;
#pragma unroll
    for (int kt = 0; kt < 4; ++kt) A[kt] = frag_from_row(xr, half, kt * 32);
#pragma unroll
    for (int kt = 4; kt < 8; ++kt) A[kt] = frag_from_row(aggm + m * HID, half, (kt - 4) * 32);
#pragma unroll
    for (int nt = 0; nt < 8; ++nt) {
      int n = (lane & 15) + nt * 16;
      const __bf16* wr = WoutB + (size_t)n * (2 * HID);
      v8f c = {};
#pragma unroll
      for (int kt = 0; kt < 8; ++kt)
        c = wmma_bf16(A[kt], frag_from_row_bf(wr, half, kt * 32), c);
      float bias = bout[n];
#pragma unroll
      for (int r = 0; r < 8; ++r) {
        int mm = r + half * 8;
        rows[mm * HID + n] = dst_x[(size_t)(row0 + mm) * HID + n] + c[r] + bias;
      }
    }
  }
  // stage 3: LayerNorm over each 128-wide row. lane & lane^16 share a row.
  {
    int rowm = lane & 15;
    int c0 = half * 64;
    float s = 0.f, s2 = 0.f;
#pragma unroll 8
    for (int j = 0; j < 64; ++j) {
      float v = rows[rowm * HID + c0 + j];
      s += v; s2 += v * v;
    }
    s  += __shfl_xor(s, 16);
    s2 += __shfl_xor(s2, 16);
    float mu  = s * (1.f / HID);
    float var = s2 * (1.f / HID) - mu * mu;
    float rstd = rsqrtf(var + 1e-5f);
#pragma unroll 8
    for (int j = 0; j < 64; ++j) {
      int col = c0 + j;
      float v = rows[rowm * HID + col];
      out[(size_t)(row0 + rowm) * HID + col] = (v - mu) * rstd * gamma[col] + beta[col];
    }
  }
}

// ---- host launcher --------------------------------------------------------
extern "C" void kernel_launch(void* const* d_in, const int* in_sizes, int n_in,
                              void* d_out, int out_size, void* d_ws, size_t ws_size,
                              hipStream_t stream) {
  const float* src_x     = (const float*)d_in[0];
  const float* dst_x     = (const float*)d_in[1];
  const int*   edge_index= (const int*)  d_in[2];
  const float* edge_attr = (const float*)d_in[3];
  const float* Wsrc  = (const float*)d_in[4];  const float* bsrc  = (const float*)d_in[5];
  const float* Wdst  = (const float*)d_in[6];  const float* bdst  = (const float*)d_in[7];
  const float* Wedge = (const float*)d_in[8];  const float* bedge = (const float*)d_in[9];
  const float* Wattn = (const float*)d_in[10]; const float* battn = (const float*)d_in[11];
  const float* Wmsg  = (const float*)d_in[12]; const float* bmsg  = (const float*)d_in[13];
  const float* Wmerge= (const float*)d_in[14]; const float* bmerge= (const float*)d_in[15];
  const float* Wout  = (const float*)d_in[16]; const float* bout  = (const float*)d_in[17];
  const float* gamma = (const float*)d_in[18]; const float* beta  = (const float*)d_in[19];

  const int Ns = in_sizes[0] / HID;     // 20000 src nodes
  const int Nd = in_sizes[1] / HID;     // 20000 dst nodes
  const int E_ = in_sizes[2] / 2;       // 320000 edges
  const int* src_idx = edge_index;
  const int* dst_idx = edge_index + E_;

  // workspace layout (~29 MB): SrcP | DstP | scores | smax | ssum | agg | bf16 weights
  float*    SrcP   = (float*)d_ws;
  float*    DstP   = SrcP + (size_t)Ns * HID;
  float*    scores = DstP + (size_t)Nd * HID;
  unsigned* smax   = (unsigned*)(scores + (size_t)E_ * NHEAD);
  float*    ssum   = (float*)(smax + (size_t)Nd * NHEAD);
  float*    agg    = ssum + (size_t)Nd * NHEAD;
  __bf16*   wb     = (__bf16*)(agg + (size_t)Nd * HD);

  // one-shot weight conversion (L2-resident, ~150 KB)
  cvt_weights_kernel<<<(WB_TOTAL + 255) / 256, 256, 0, stream>>>(Wsrc, Wdst, Wedge, Wmsg,
                                                                 Wmerge, Wout, wb);

  // zero smax (enc=0 is the minimum of the order-preserving encoding), ssum, agg
  hipMemsetAsync(smax, 0, (size_t)(Nd * NHEAD * 2 + Nd * HD) * sizeof(float), stream);

  int tiles_s = (Ns + 15) / 16;
  int tiles_d = (Nd + 15) / 16;
  proj_kernel<<<(tiles_s + 7) / 8, 256, 0, stream>>>(src_x, wb + OFF_WSRC, bsrc, SrcP, Ns);
  proj_kernel<<<(tiles_d + 7) / 8, 256, 0, stream>>>(dst_x, wb + OFF_WDST, bdst, DstP, Nd);

  int etiles = (E_ + 15) / 16;
  edge_scores_kernel<<<(etiles + WPB - 1) / WPB, WPB * 32, 0, stream>>>(
      edge_attr, wb + OFF_WEDGE, bedge, Wattn, battn, src_idx, dst_idx, SrcP, DstP,
      scores, smax, E_);

  edge_exp_kernel<<<(E_ * NHEAD + 255) / 256, 256, 0, stream>>>(dst_idx, scores, smax, ssum,
                                                                E_ * NHEAD);

  edge_aggregate_kernel<<<(etiles + WPB - 1) / WPB, WPB * 32, 0, stream>>>(
      edge_attr, wb + OFF_WEDGE, bedge, wb + OFF_WMSG, bmsg, src_idx, dst_idx, SrcP,
      scores, ssum, agg, E_);

  finalize_kernel<<<(tiles_d + 1) / 2, 64, 0, stream>>>(dst_x, agg, wb + OFF_WMERGE, bmerge,
                                                        wb + OFF_WOUT, bout, gamma, beta,
                                                        (float*)d_out, Nd);
}
